// TG_MSA_57294863729441
// MI455X (gfx1250) — matrix operand order repaired and checked by reading, not compiled, not run on hardware
//
#include <hip/hip_runtime.h>

// ---------------------------------------------------------------------------
// Problem constants (from reference): B=8, C=512, H=W=32 -> N=1024,
// HEADS=8, DHEAD=64, EPS=0.05, ITERS=100.
// ---------------------------------------------------------------------------
#define BB     8
#define CC     512
#define NN     1024
#define HEADS  8
#define DH     64
#define EPSV   0.05f
#define INVEPS 20.0f
#define ITERS  100

typedef __bf16 bf16;
typedef __attribute__((ext_vector_type(16))) __bf16 v16bf;
typedef __attribute__((ext_vector_type(8)))  __bf16 v8bf;
typedef __attribute__((ext_vector_type(8)))  float  v8f;

// LDS byte address of a generic pointer to __shared__ (flat LDS addr bits[31:0])
__device__ inline unsigned lds_u32(const void* p) {
  return (unsigned)(unsigned long long)(uintptr_t)p;
}

// ---------------------------------------------------------------------------
// Small utility kernels
// ---------------------------------------------------------------------------
__global__ void k_zero(float* p, int n) {
  int i = blockIdx.x * blockDim.x + threadIdx.x;
  if (i < n) p[i] = 0.0f;
}

// f32 [bz][R][Cc] -> bf16 [bz][Cc][R]  (transpose + convert)
__global__ void k_t2b(const float* __restrict__ in, bf16* __restrict__ out,
                      int R, int Cc, long sIn, long sOut) {
  int bz = blockIdx.y;
  size_t idx = (size_t)blockIdx.x * blockDim.x + threadIdx.x;
  if (idx >= (size_t)R * Cc) return;
  int r = (int)(idx / Cc), c = (int)(idx % Cc);
  out[(size_t)bz * sOut + (size_t)c * R + r] = (bf16)in[(size_t)bz * sIn + idx];
}

// straight f32 -> bf16 convert
__global__ void k_c2b(const float* __restrict__ in, bf16* __restrict__ out, size_t total) {
  size_t i = (size_t)blockIdx.x * blockDim.x + threadIdx.x;
  if (i < total) out[i] = (bf16)in[i];
}

// column squared-norms of [bz][C][N] f32 over C -> out[bz][N]  (t row norms)
__global__ void k_colnorm(const float* __restrict__ in, float* __restrict__ out) {
  int n = blockIdx.x * blockDim.x + threadIdx.x;
  int bz = blockIdx.y;
  const float* p = in + (size_t)bz * CC * NN;
  float s = 0.0f;
  for (int c = 0; c < CC; ++c) { float x = p[(size_t)c * NN + n]; s += x * x; }
  out[(size_t)bz * NN + n] = s;
}

// row squared-norms of [rows][C] f32 (wave per row)
__global__ __launch_bounds__(256)
void k_rownorm(const float* __restrict__ in, float* __restrict__ out, int C) {
  int lane = threadIdx.x & 31, w = threadIdx.x >> 5;
  size_t row = (size_t)blockIdx.x * 8 + w;
  const float* p = in + row * C;
  float s = 0.0f;
  for (int c = lane; c < C; c += 32) { float x = p[c]; s += x * x; }
  for (int off = 16; off > 0; off >>= 1) s += __shfl_xor(s, off);
  if (lane == 0) out[row] = s;
}

// ---------------------------------------------------------------------------
// Generic bf16 WMMA GEMM:  C[M,N] = A[M,K] @ B[K,N], B given as BT[N][K].
// Block = 256 threads = 8 waves arranged 4x2; each wave computes a 32x64 tile
// (2 A-frags x 4 B-frags -> 8 v_wmma per K-step, ~21 FLOP/B register reuse).
// Block tile = 128x128. Fragment layouts per CDNA5 ISA 7.12.2.
// ---------------------------------------------------------------------------
enum { EP_PLAIN = 0, EP_COST = 1, EP_GUIDED = 2, EP_BIAS = 3, EP_GELU = 4, EP_FINAL = 5 };

__device__ inline v16bf mk16(const bf16* __restrict__ p) {
  // A frag halves: lanes 0-15 hold K{0..7,16..23}, lanes 16-31 K{8..15,24..31}
  v8bf lo = *(const v8bf*)p;
  v8bf hi = *(const v8bf*)(p + 16);
  v16bf r;
#pragma unroll
  for (int i = 0; i < 8; ++i) { r[i] = lo[i]; r[i + 8] = hi[i]; }
  return r;
}

template <int EP>
__global__ __launch_bounds__(256)
void k_gemm(const bf16* __restrict__ A, long sA, int lda,
            const bf16* __restrict__ BT, long sB, int ldb,
            float* outF, long sF, int ldf,
            bf16* outB, long sOB, int ldob,
            const float* __restrict__ ex0, long sE0,
            const float* __restrict__ ex1, long sE1,
            int M, int N, int K) {
  const int bz = blockIdx.z;
  const int lane = threadIdx.x & 31;
  const int wid  = threadIdx.x >> 5;
  const int row0 = blockIdx.y * 128 + (wid >> 1) * 32;
  const int col0 = blockIdx.x * 128 + (wid & 1) * 64;
  if (row0 >= M || col0 >= N) return;     // wave-uniform: EXEC stays all-ones
  const int l15 = lane & 15, lhi = lane >> 4;

  const bf16* a0p = A + (size_t)bz * sA + (size_t)(row0 + l15) * lda + lhi * 8;
  const bf16* a1p = a0p + (size_t)16 * lda;
  // B frag: lane(0-15) col n=l15 holds K{0..7,16..23}; lane(16-31) K{8..15,24..31}
  const bf16* bp0 = BT + (size_t)bz * sB + (size_t)(col0 + l15) * ldb + lhi * 8;
  const size_t bStep = (size_t)16 * ldb;

  v8f acc[8];
#pragma unroll
  for (int i = 0; i < 8; ++i) acc[i] = (v8f){};

  for (int kk = 0; kk < K; kk += 32) {
    __builtin_prefetch(a0p + kk + 64, 0, 1);
    __builtin_prefetch(bp0 + kk + 64, 0, 1);
    v16bf a0 = mk16(a0p + kk);
    v16bf a1 = mk16(a1p + kk);
    v16bf b0 = mk16(bp0 + kk);
    v16bf b1 = mk16(bp0 + bStep + kk);
    v16bf b2 = mk16(bp0 + 2 * bStep + kk);
    v16bf b3 = mk16(bp0 + 3 * bStep + kk);
    acc[0] = __builtin_amdgcn_wmma_f32_16x16x32_bf16(false, a0, false, b0, (short)0, acc[0], false, false);
    acc[1] = __builtin_amdgcn_wmma_f32_16x16x32_bf16(false, a0, false, b1, (short)0, acc[1], false, false);
    acc[2] = __builtin_amdgcn_wmma_f32_16x16x32_bf16(false, a0, false, b2, (short)0, acc[2], false, false);
    acc[3] = __builtin_amdgcn_wmma_f32_16x16x32_bf16(false, a0, false, b3, (short)0, acc[3], false, false);
    acc[4] = __builtin_amdgcn_wmma_f32_16x16x32_bf16(false, a1, false, b0, (short)0, acc[4], false, false);
    acc[5] = __builtin_amdgcn_wmma_f32_16x16x32_bf16(false, a1, false, b1, (short)0, acc[5], false, false);
    acc[6] = __builtin_amdgcn_wmma_f32_16x16x32_bf16(false, a1, false, b2, (short)0, acc[6], false, false);
    acc[7] = __builtin_amdgcn_wmma_f32_16x16x32_bf16(false, a1, false, b3, (short)0, acc[7], false, false);
  }

  // C/D layout: VGPR i -> lanes 0-15 row=base+i, lanes 16-31 row=base+i+8; col=base+l15
#pragma unroll
  for (int rr = 0; rr < 2; ++rr) {
#pragma unroll
    for (int cc2 = 0; cc2 < 4; ++cc2) {
#pragma unroll
      for (int i = 0; i < 8; ++i) {
        int m = row0 + rr * 16 + i + lhi * 8;
        int c = col0 + cc2 * 16 + l15;
        float v = acc[rr * 4 + cc2][i];
        if (EP == EP_COST) {
          // squared-distance cost: |t_m|^2 + |v_c|^2 - 2 <t_m, v_c>
          v = ex0[(size_t)bz * sE0 + m] + ex1[(size_t)bz * sE1 + c] - 2.0f * v;
          outF[(size_t)bz * sF + (size_t)m * ldf + c] = v;
        } else if (EP == EP_GUIDED) {
          v *= ex0[(size_t)bz * sE0 + (size_t)m * ldob + c];  // elementwise * v_inp
          outB[(size_t)bz * sOB + (size_t)m * ldob + c] = (bf16)v;
        } else if (EP == EP_BIAS) {
          v += ex0[c];
          outF[(size_t)bz * sF + (size_t)m * ldf + c] = v;
        } else if (EP == EP_GELU) {
          v += ex0[c];
          v = 0.5f * v * (1.0f + erff(v * 0.70710678118f));  // exact gelu
          outB[(size_t)bz * sOB + (size_t)m * ldob + c] = (bf16)v;
        } else if (EP == EP_FINAL) {
          // out = acc + b2[c] + out_c[m][c]; store transposed to [bz][c][m]
          v += ex0[c] + ex1[(size_t)bz * sE1 + (size_t)m * ldf + c];
          outF[(size_t)bz * sF + (size_t)c * M + m] = v;
        } else {  // EP_PLAIN
          if (outF) outF[(size_t)bz * sF + (size_t)m * ldf + c] = v;
          if (outB) outB[(size_t)bz * sOB + (size_t)m * ldob + c] = (bf16)v;
        }
      }
    }
  }
}

// ---------------------------------------------------------------------------
// Cost-matrix statistics (mean / var) via block partials + float atomics
// ---------------------------------------------------------------------------
__global__ __launch_bounds__(256)
void k_msum(const float* __restrict__ Mraw, float* __restrict__ stats) {
  __shared__ float s1[256], s2[256];
  int bz = blockIdx.y;
  const float* p = Mraw + (size_t)bz * (NN * NN);
  size_t base = (size_t)blockIdx.x * 256 + threadIdx.x;
  float a = 0.0f, b = 0.0f;
  for (int t = 0; t < 4; ++t) {
    float x = p[base + (size_t)t * 262144];
    a += x; b += x * x;
  }
  s1[threadIdx.x] = a; s2[threadIdx.x] = b;
  __syncthreads();
  for (int o = 128; o > 0; o >>= 1) {
    if ((int)threadIdx.x < o) {
      s1[threadIdx.x] += s1[threadIdx.x + o];
      s2[threadIdx.x] += s2[threadIdx.x + o];
    }
    __syncthreads();
  }
  if (threadIdx.x == 0) {
    atomicAdd(&stats[bz * 2 + 0], s1[0]);
    atomicAdd(&stats[bz * 2 + 1], s2[0]);
  }
}

// standardize M in place and write transposed copy MT (both L2-resident: 64MB << 192MB)
__global__ void k_mnorm(float* __restrict__ Mb, float* __restrict__ MTb,
                        const float* __restrict__ stats) {
  int bz = blockIdx.y;
  size_t idx = (size_t)blockIdx.x * 256 + threadIdx.x;
  int r = (int)(idx >> 10), c = (int)(idx & 1023);
  float mean = stats[bz * 2 + 0] * (1.0f / 1048576.0f);
  float var  = stats[bz * 2 + 1] * (1.0f / 1048576.0f) - mean * mean;
  float rstd = rsqrtf(fmaxf(var, 1e-20f));
  size_t o = (size_t)bz * (NN * NN);
  float v = (Mb[o + idx] - mean) * rstd;
  Mb[o + idx] = v;
  MTb[o + (size_t)c * NN + r] = v;
}

// ---------------------------------------------------------------------------
// Log-domain Sinkhorn: one workgroup (32 waves) per batch; f/g live in LDS;
// each half-iteration streams M (resp. MT) row-major; online logsumexp per
// lane + __shfl_xor wave merge. 100 iterations, workgroup barriers between.
// ---------------------------------------------------------------------------
__global__ __launch_bounds__(1024)
void k_sinkhorn(const float* __restrict__ Mn, const float* __restrict__ MTn,
                float* __restrict__ fg, int iters) {
  __shared__ float fs[NN], gs[NN];
  const int bz = blockIdx.x;
  const int tid = threadIdx.x;
  const int lane = tid & 31, w = tid >> 5;
  fs[tid] = 0.0f; gs[tid] = 0.0f;
  __syncthreads();
  const float log_mu = -__logf(1024.0f);
  const float* Mb  = Mn  + (size_t)bz * (NN * NN);
  const float* MTb = MTn + (size_t)bz * (NN * NN);

  for (int it = 0; it < iters; ++it) {
    // f[i] = eps*(log_mu - lse_j((g[j]-M[i][j])/eps))
    for (int j = 0; j < 32; ++j) {
      int r = w * 32 + j;
      const float* row = Mb + (size_t)r * NN;
      float m = -3.0e38f, s = 0.0f;
      for (int t = 0; t < 32; ++t) {
        int c = lane + t * 32;
        float x = (gs[c] - row[c]) * INVEPS;
        if (x > m) { s = s * __expf(m - x) + 1.0f; m = x; }
        else       { s += __expf(x - m); }
      }
      for (int off = 16; off > 0; off >>= 1) {
        float om = __shfl_xor(m, off);
        float os = __shfl_xor(s, off);
        if (om > m) { s = s * __expf(m - om) + os; m = om; }
        else        { s += os * __expf(om - m); }
      }
      if (lane == 0) fs[r] = EPSV * (log_mu - (m + __logf(s)));
    }
    __syncthreads();
    // g[j] = eps*(log_mu - lse_i((f[i]-M[i][j])/eps))  via MT rows
    for (int j = 0; j < 32; ++j) {
      int r = w * 32 + j;
      const float* row = MTb + (size_t)r * NN;
      float m = -3.0e38f, s = 0.0f;
      for (int t = 0; t < 32; ++t) {
        int c = lane + t * 32;
        float x = (fs[c] - row[c]) * INVEPS;
        if (x > m) { s = s * __expf(m - x) + 1.0f; m = x; }
        else       { s += __expf(x - m); }
      }
      for (int off = 16; off > 0; off >>= 1) {
        float om = __shfl_xor(m, off);
        float os = __shfl_xor(s, off);
        if (om > m) { s = s * __expf(m - om) + os; m = om; }
        else        { s += os * __expf(om - m); }
      }
      if (lane == 0) gs[r] = EPSV * (log_mu - (m + __logf(s)));
    }
    __syncthreads();
  }
  fg[(size_t)bz * 2048 + tid] = fs[tid];
  fg[(size_t)bz * 2048 + 1024 + tid] = gs[tid];
}

// Ges = exp((f[i]+g[j]-M)/eps)  -> bf16 (A operand of guided GEMM)
__global__ void k_plan(const float* __restrict__ Mn, const float* __restrict__ fg,
                       bf16* __restrict__ Ges) {
  int bz = blockIdx.y;
  size_t idx = (size_t)blockIdx.x * 256 + threadIdx.x;
  int r = (int)(idx >> 10), c = (int)(idx & 1023);
  float f = fg[(size_t)bz * 2048 + r];
  float g = fg[(size_t)bz * 2048 + 1024 + c];
  float m = Mn[(size_t)bz * (NN * NN) + idx];
  Ges[(size_t)bz * (NN * NN) + idx] = (bf16)__expf((f + g - m) * INVEPS);
}

// ---------------------------------------------------------------------------
// Channel attention per (b,h): column L2-norms of q/k head block, 64x64 Gram
// matrix over n=1024. Chunks of 64 rows are staged into LDS with
// GLOBAL_LOAD_ASYNC_TO_LDS_B128 (ASYNCcnt; s_wait_asynccnt before barrier),
// then rescale + softmax; writes bf16 attn[b][h][d][e] (BT of the xo GEMM).
// ---------------------------------------------------------------------------
__global__ __launch_bounds__(256)
void k_attn(const float* __restrict__ q, const float* __restrict__ k,
            const float* __restrict__ rescale, bf16* __restrict__ attn) {
  __shared__ float ks[64][64];   // dense: async b128 writes 16B-aligned rows
  __shared__ float qs[64][64];
  __shared__ float att[64][65];
  __shared__ float kn[64], qn[64];
  const int bh = blockIdx.x;
  const int bz = bh >> 3, h = bh & 7;
  const int tid = threadIdx.x;
  const float* kb = k + (size_t)bz * NN * CC + h * DH;
  const float* qb = q + (size_t)bz * NN * CC + h * DH;

  if (tid < 64) {
    float s = 0.0f;
    for (int nn2 = 0; nn2 < NN; ++nn2) { float x = kb[(size_t)nn2 * CC + tid]; s += x * x; }
    kn[tid] = fmaxf(sqrtf(s), 1e-12f);
  } else if (tid < 128) {
    int d = tid - 64; float s = 0.0f;
    for (int nn2 = 0; nn2 < NN; ++nn2) { float x = qb[(size_t)nn2 * CC + d]; s += x * x; }
    qn[d] = fmaxf(sqrtf(s), 1e-12f);
  }

  float acc[16];
#pragma unroll
  for (int t = 0; t < 16; ++t) acc[t] = 0.0f;

  for (int c0 = 0; c0 < NN; c0 += 64) {
    __syncthreads();  // previous chunk fully consumed before overwrite
#pragma unroll
    for (int t = 0; t < 4; ++t) {
      int fo = (t * 256 + tid) * 4;         // float offset within 64x64 tile
      int r = fo >> 6, d = fo & 63;
      unsigned lk = lds_u32(&ks[r][d]);
      unsigned lq = lds_u32(&qs[r][d]);
      unsigned long long gk = (unsigned long long)(uintptr_t)(kb + (size_t)(c0 + r) * CC + d);
      unsigned long long gq = (unsigned long long)(uintptr_t)(qb + (size_t)(c0 + r) * CC + d);
      asm volatile("global_load_async_to_lds_b128 %0, %1, off"
                   :: "v"(lk), "v"(gk) : "memory");
      asm volatile("global_load_async_to_lds_b128 %0, %1, off"
                   :: "v"(lq), "v"(gq) : "memory");
    }
    asm volatile("s_wait_asynccnt 0x0" ::: "memory");
    __syncthreads();
#pragma unroll 4
    for (int r = 0; r < 64; ++r) {
#pragma unroll
      for (int t = 0; t < 16; ++t) {
        int idx = tid + 256 * t;
        int d = idx >> 6, e = idx & 63;
        acc[t] += ks[r][d] * qs[r][e];
      }
    }
  }
  __syncthreads();
  float rs = rescale[h];
#pragma unroll
  for (int t = 0; t < 16; ++t) {
    int idx = tid + 256 * t;
    int d = idx >> 6, e = idx & 63;
    att[d][e] = rs * acc[t] / (kn[d] * qn[e]);
  }
  __syncthreads();
  if (tid < 64) {
    float mx = -3.0e38f;
    for (int e = 0; e < 64; ++e) mx = fmaxf(mx, att[tid][e]);
    float sum = 0.0f;
    for (int e = 0; e < 64; ++e) { float ex = __expf(att[tid][e] - mx); att[tid][e] = ex; sum += ex; }
    float inv = 1.0f / sum;
    bf16* ap = attn + ((size_t)bh * 64 + tid) * 64;
    for (int e = 0; e < 64; ++e) ap[e] = (bf16)(att[tid][e] * inv);
  }
}

// ---------------------------------------------------------------------------
// Host-side orchestration
// ---------------------------------------------------------------------------
extern "C" void kernel_launch(void* const* d_in, const int* in_sizes, int n_in,
                              void* d_out, int out_size, void* d_ws, size_t ws_size,
                              hipStream_t stream) {
  const float* x_in   = (const float*)d_in[0];
  const float* task_x = (const float*)d_in[1];
  const float* Wq     = (const float*)d_in[2];
  const float* Wk     = (const float*)d_in[3];
  const float* Wv     = (const float*)d_in[4];
  const float* rescale= (const float*)d_in[5];
  const float* Wp     = (const float*)d_in[6];
  const float* bp     = (const float*)d_in[7];
  const float* W1     = (const float*)d_in[8];
  const float* b1     = (const float*)d_in[9];
  const float* W2     = (const float*)d_in[10];
  const float* b2     = (const float*)d_in[11];

  size_t off = 0;
  auto alloc = [&](size_t bytes) -> void* {
    void* p = (char*)d_ws + off;
    off = (off + bytes + 255) & ~(size_t)255;
    return p;
  };

  float* stats = (float*)alloc(BB * 2 * sizeof(float));
  bf16* xb  = (bf16*)alloc((size_t)BB * NN * CC * 2);   // x   [b][n][c]  bf16
  bf16* tb  = (bf16*)alloc((size_t)BB * NN * CC * 2);   // t   [b][n][c]  bf16
  bf16* tT  = (bf16*)alloc((size_t)BB * CC * NN * 2);   // t^T [b][c][n]  bf16
  bf16* WqT = (bf16*)alloc((size_t)CC * CC * 2);
  bf16* WkT = (bf16*)alloc((size_t)CC * CC * 2);
  bf16* WvT = (bf16*)alloc((size_t)CC * CC * 2);
  bf16* W1T = (bf16*)alloc((size_t)CC * CC * 2);
  bf16* W2T = (bf16*)alloc((size_t)CC * CC * 2);
  bf16* WpT = (bf16*)alloc((size_t)CC * CC * 2);
  float* tnorm = (float*)alloc((size_t)BB * NN * 4);
  float* vnorm = (float*)alloc((size_t)BB * NN * 4);
  float* qf = (float*)alloc((size_t)BB * NN * CC * 4);
  float* kf = (float*)alloc((size_t)BB * NN * CC * 4);
  float* vf = (float*)alloc((size_t)BB * NN * CC * 4);
  bf16* vb  = (bf16*)alloc((size_t)BB * NN * CC * 2);
  float* Mb  = (float*)alloc((size_t)BB * NN * NN * 4);
  float* MTb = (float*)alloc((size_t)BB * NN * NN * 4);
  float* fg  = (float*)alloc((size_t)BB * 2 * NN * 4);
  bf16* Ges  = (bf16*)alloc((size_t)BB * NN * NN * 2);
  bf16* gdd  = (bf16*)alloc((size_t)BB * NN * CC * 2); // guided bf16
  bf16* attnb= (bf16*)alloc((size_t)BB * HEADS * DH * DH * 2);
  bf16* xo   = (bf16*)alloc((size_t)BB * NN * CC * 2);
  float* outc= (float*)alloc((size_t)BB * NN * CC * 4);
  bf16* a1   = (bf16*)alloc((size_t)BB * NN * CC * 2);
  float* outp = (float*)d_out;

  const long sNC = (long)NN * CC;
  const long sNN = (long)NN * NN;

  k_zero<<<1, 64, 0, stream>>>(stats, BB * 2);

  // packing / transposes (x_in,task_x are [b][C][N]; x/t row-major is their transpose)
  k_t2b<<<dim3((CC * NN + 255) / 256, BB), 256, 0, stream>>>(x_in,   xb, CC, NN, (long)CC * NN, sNC);
  k_t2b<<<dim3((CC * NN + 255) / 256, BB), 256, 0, stream>>>(task_x, tb, CC, NN, (long)CC * NN, sNC);
  k_c2b<<<( (size_t)BB * CC * NN + 255) / 256, 256, 0, stream>>>(task_x, tT, (size_t)BB * CC * NN);
  k_t2b<<<dim3((CC * CC + 255) / 256, 1), 256, 0, stream>>>(Wq, WqT, CC, CC, 0, 0);
  k_t2b<<<dim3((CC * CC + 255) / 256, 1), 256, 0, stream>>>(Wk, WkT, CC, CC, 0, 0);
  k_t2b<<<dim3((CC * CC + 255) / 256, 1), 256, 0, stream>>>(Wv, WvT, CC, CC, 0, 0);
  k_t2b<<<dim3((CC * CC + 255) / 256, 1), 256, 0, stream>>>(W1, W1T, CC, CC, 0, 0);
  k_t2b<<<dim3((CC * CC + 255) / 256, 1), 256, 0, stream>>>(W2, W2T, CC, CC, 0, 0);
  k_t2b<<<dim3((CC * CC + 255) / 256, 1), 256, 0, stream>>>(Wp, WpT, CC, CC, 0, 0);

  k_colnorm<<<dim3(NN / 256, BB), 256, 0, stream>>>(task_x, tnorm);

  // q, k, v GEMMs: [1024x512] = x @ W  (WMMA bf16, 128x128 block tiles)
  dim3 gQ(CC / 128, NN / 128, BB);
  k_gemm<EP_PLAIN><<<gQ, 256, 0, stream>>>(xb, sNC, CC, WqT, 0, CC, qf, sNC, CC,
                                           nullptr, 0, 0, nullptr, 0, nullptr, 0, NN, CC, CC);
  k_gemm<EP_PLAIN><<<gQ, 256, 0, stream>>>(xb, sNC, CC, WkT, 0, CC, kf, sNC, CC,
                                           nullptr, 0, 0, nullptr, 0, nullptr, 0, NN, CC, CC);
  k_gemm<EP_PLAIN><<<gQ, 256, 0, stream>>>(xb, sNC, CC, WvT, 0, CC, vf, sNC, CC,
                                           vb, sNC, CC, nullptr, 0, nullptr, 0, NN, CC, CC);

  k_rownorm<<<(BB * NN) / 8, 256, 0, stream>>>(vf, vnorm, CC);

  // cost matrix M = |t|^2 + |v|^2 - 2 t@v^T   (v row-major bf16 is exactly BT)
  dim3 gM(NN / 128, NN / 128, BB);
  k_gemm<EP_COST><<<gM, 256, 0, stream>>>(tb, sNC, CC, vb, sNC, CC, Mb, sNN, NN,
                                          nullptr, 0, 0, tnorm, NN, vnorm, NN, NN, NN, CC);

  k_msum<<<dim3(1024, BB), 256, 0, stream>>>(Mb, stats);
  k_mnorm<<<dim3(4096, BB), 256, 0, stream>>>(Mb, MTb, stats);

  k_sinkhorn<<<BB, 1024, 0, stream>>>(Mb, MTb, fg, ITERS);
  k_plan<<<dim3(4096, BB), 256, 0, stream>>>(Mb, fg, Ges);

  // guided = (Ges @ t) * v   (A = Ges bf16, BT = t^T bf16, epilogue *v)
  k_gemm<EP_GUIDED><<<gQ, 256, 0, stream>>>(Ges, sNN, NN, tT, (long)CC * NN, NN,
                                            nullptr, 0, 0, gdd, sNC, CC,
                                            vf, sNC, nullptr, 0, NN, CC, NN);

  // channel attention (norms + Gram + softmax), writes bf16 attn[b][h][64][64]
  k_attn<<<BB * HEADS, 256, 0, stream>>>(qf, kf, rescale, attnb);

  // xo per head: xo[:,h*64+d] = sum_e attn[h][d][e] * guided[:,h*64+e]
  for (int h = 0; h < HEADS; ++h) {
    k_gemm<EP_PLAIN><<<dim3(1, NN / 128, BB), 256, 0, stream>>>(
        gdd + h * DH, sNC, CC, attnb + (size_t)h * DH * DH, (long)HEADS * DH * DH, DH,
        nullptr, 0, 0, xo + h * DH, sNC, CC,
        nullptr, 0, nullptr, 0, NN, DH, DH);
  }

  // out_c = xo @ Wp + bp
  k_gemm<EP_BIAS><<<gQ, 256, 0, stream>>>(xo, sNC, CC, WpT, 0, CC, outc, sNC, CC,
                                          nullptr, 0, 0, bp, 0, nullptr, 0, NN, CC, CC);
  // a1 = gelu(v @ W1 + b1)  (exact gelu)
  k_gemm<EP_GELU><<<gQ, 256, 0, stream>>>(vb, sNC, CC, W1T, 0, CC, nullptr, 0, 0,
                                          a1, sNC, CC, b1, 0, nullptr, 0, NN, CC, CC);
  // out = (a1 @ W2 + b2 + out_c) transposed to [b][c][h][w]
  k_gemm<EP_FINAL><<<gQ, 256, 0, stream>>>(a1, sNC, CC, W2T, 0, CC,
                                           outp, (long)CC * NN, CC /* ex1 row stride */,
                                           nullptr, 0, 0, b2, 0, outc, sNC, NN, CC, CC);

  (void)in_sizes; (void)n_in; (void)out_size; (void)ws_size;
}
